// SimilarityComputer_68247030333454
// MI455X (gfx1250) — matrix operand order repaired and checked by reading, not compiled, not run on hardware
//
#include <hip/hip_runtime.h>
#include <hip/hip_bf16.h>
#include <math.h>

// Problem constants (match reference)
#define NROW 8192
#define DDIM 512
#define KTOP 49            // k-1 = 49 off-diagonal neighbors

typedef __attribute__((ext_vector_type(16))) __bf16 v16bf;
typedef __attribute__((ext_vector_type(8)))  float  v8f;
typedef int v4i __attribute__((vector_size(16)));   // matches builtin param type

union FragAB { v16bf v; uint4 q[2]; };

#define AS1 __attribute__((address_space(1)))
#define AS3 __attribute__((address_space(3)))

#if defined(__has_builtin)
#  if __has_builtin(__builtin_amdgcn_global_load_async_to_lds_b128)
#    define HAVE_ASYNC_LDS 1
#  endif
#endif

__device__ __forceinline__ void wait_async_all() {
#if defined(__has_builtin) && __has_builtin(__builtin_amdgcn_s_wait_asynccnt)
    __builtin_amdgcn_s_wait_asynccnt(0);
#else
    asm volatile("s_wait_asynccnt 0x0" ::: "memory");
#endif
}

__device__ __forceinline__ unsigned short f2bf(float x) {
    union { float f; unsigned int u; } c; c.f = x;
    unsigned int u = c.u;
    unsigned int r = u + 0x7FFFu + ((u >> 16) & 1u);  // round-to-nearest-even
    return (unsigned short)(r >> 16);
}

// ---------------------------------------------------------------------------
// 1) softmax(similarity_weights) and blend 4 weight matrices + biases.
// ---------------------------------------------------------------------------
__global__ __launch_bounds__(256)
void fuse_weights_kernel(const float* __restrict__ sw,
                         const float* __restrict__ W0, const float* __restrict__ b0,
                         const float* __restrict__ W1, const float* __restrict__ b1,
                         const float* __restrict__ W2, const float* __restrict__ b2,
                         const float* __restrict__ W3, const float* __restrict__ b3,
                         unsigned short* __restrict__ Wf, float* __restrict__ bf_) {
    float s0 = sw[0], s1 = sw[1], s2 = sw[2], s3 = sw[3];
    float m  = fmaxf(fmaxf(s0, s1), fmaxf(s2, s3));
    float e0 = expf(s0 - m), e1 = expf(s1 - m), e2 = expf(s2 - m), e3 = expf(s3 - m);
    float inv = 1.0f / (e0 + e1 + e2 + e3);
    float w0 = e0 * inv, w1 = e1 * inv, w2 = e2 * inv, w3 = e3 * inv;

    int gid = blockIdx.x * 256 + threadIdx.x;
    if (gid < DDIM * DDIM)
        Wf[gid] = f2bf(w0 * W0[gid] + w1 * W1[gid] + w2 * W2[gid] + w3 * W3[gid]);
    if (gid < DDIM)
        bf_[gid] = w0 * b0[gid] + w1 * b1[gid] + w2 * b2[gid] + w3 * b3[gid];
}

// ---------------------------------------------------------------------------
// 2) f32 -> bf16 conversion
// ---------------------------------------------------------------------------
__global__ __launch_bounds__(256)
void f32_to_bf16_kernel(const float* __restrict__ src, unsigned short* __restrict__ dst, long n) {
    long gid = (long)blockIdx.x * 256 + threadIdx.x;
    if (gid < n) dst[gid] = f2bf(src[gid]);
}

// ---------------------------------------------------------------------------
// Generic WMMA GEMM:  C[M][N] = A[M][K] * Bt[N][K]^T  (+ bias[N] optional)
// Block tile 128(M) x 128(N); 256 threads = 8 waves in 4(M) x 2(N);
// each wave computes a 32x64 patch = 2x4 wmma tiles; K stepped by 32.
// Tiles stream to LDS via GLOBAL_LOAD_ASYNC_TO_LDS_B128 with double buffering
// (fallback: register-prefetch software pipeline). M,N % 128 == 0, K % 32 == 0.
// ---------------------------------------------------------------------------
#define LSTR 40   // LDS row stride in halves (32 data + 8 pad)

__global__ __launch_bounds__(256)
void wmma_gemm_abt_kernel(const unsigned short* __restrict__ A,
                          const unsigned short* __restrict__ Bt,
                          const float* __restrict__ bias,
                          float* __restrict__ C,
                          int M, int N, int K, int addBias) {
    __shared__ unsigned short As[2][128 * LSTR];
    __shared__ unsigned short Bs[2][128 * LSTR];

    const int tid   = threadIdx.x;
    const int wave  = tid >> 5;
    const int lane  = tid & 31;
    const int waveM = wave & 3;               // 0..3  -> 32-row slab
    const int waveN = wave >> 2;              // 0..1  -> 64-col slab
    const int m0    = blockIdx.y * 128;
    const int n0    = blockIdx.x * 128;

    const int lhalf = lane >> 4;              // 0 or 1
    const int l15   = lane & 15;

    // Uniform tile-load mapping: A tile 128x32 = 512 x 16B chunks, 2/thread;
    // B tile likewise. chunk c: row = c>>2, seg = c&3.
    const int r0 = tid >> 2, sg = tid & 3;    // chunk tid      -> rows 0..63
    const int r1 = r0 + 64;                   // chunk tid+256  -> rows 64..127
    const int ldsOff = r0 * LSTR + sg * 8;    // halves
    const int ldsOff2 = r1 * LSTR + sg * 8;

    const int nsteps = K >> 5;

    v8f acc[2][4];
    #pragma unroll
    for (int i = 0; i < 2; ++i)
        #pragma unroll
        for (int j = 0; j < 4; ++j)
            acc[i][j] = (v8f){0.f,0.f,0.f,0.f,0.f,0.f,0.f,0.f};

#if defined(HAVE_ASYNC_LDS)
    // ---- async double-buffered pipeline --------------------------------
    #define ISSUE_TILE(buf, kk)                                                        \
        do {                                                                           \
            __builtin_amdgcn_global_load_async_to_lds_b128(                            \
                (AS1 v4i*)&A[(size_t)(m0 + r0) * K + (kk) + sg * 8],                   \
                (AS3 v4i*)&As[(buf)][ldsOff], 0, 0);                                   \
            __builtin_amdgcn_global_load_async_to_lds_b128(                            \
                (AS1 v4i*)&A[(size_t)(m0 + r1) * K + (kk) + sg * 8],                   \
                (AS3 v4i*)&As[(buf)][ldsOff2], 0, 0);                                  \
            __builtin_amdgcn_global_load_async_to_lds_b128(                            \
                (AS1 v4i*)&Bt[(size_t)(n0 + r0) * K + (kk) + sg * 8],                  \
                (AS3 v4i*)&Bs[(buf)][ldsOff], 0, 0);                                   \
            __builtin_amdgcn_global_load_async_to_lds_b128(                            \
                (AS1 v4i*)&Bt[(size_t)(n0 + r1) * K + (kk) + sg * 8],                  \
                (AS3 v4i*)&Bs[(buf)][ldsOff2], 0, 0);                                  \
        } while (0)

    ISSUE_TILE(0, 0);
    for (int i = 0; i < nsteps; ++i) {
        const int cur = i & 1;
        wait_async_all();
        __syncthreads();                       // buf[cur] filled, prev reads done
        if (i + 1 < nsteps) ISSUE_TILE(cur ^ 1, (i + 1) << 5);

        FragAB afr[2], bfr[4];
        #pragma unroll
        for (int im = 0; im < 2; ++im) {
            int row = waveM * 32 + im * 16 + l15;
            afr[im].q[0] = *(const uint4*)&As[cur][row * LSTR + lhalf * 8];
            afr[im].q[1] = *(const uint4*)&As[cur][row * LSTR + 16 + lhalf * 8];
        }
        #pragma unroll
        for (int in = 0; in < 4; ++in) {
            int col = waveN * 64 + in * 16 + l15;
            bfr[in].q[0] = *(const uint4*)&Bs[cur][col * LSTR + lhalf * 8];
            bfr[in].q[1] = *(const uint4*)&Bs[cur][col * LSTR + 16 + lhalf * 8];
        }
        #pragma unroll
        for (int im = 0; im < 2; ++im)
            #pragma unroll
            for (int in = 0; in < 4; ++in)
                acc[im][in] = __builtin_amdgcn_wmma_f32_16x16x32_bf16(
                    false, afr[im].v, false, bfr[in].v,
                    (short)0, acc[im][in], false, false);
    }
    #undef ISSUE_TILE
#else
    // ---- fallback: register-prefetch software pipeline ------------------
    uint4 ra0, ra1, rb0, rb1;
    ra0 = *(const uint4*)&A[(size_t)(m0 + r0) * K + sg * 8];
    ra1 = *(const uint4*)&A[(size_t)(m0 + r1) * K + sg * 8];
    rb0 = *(const uint4*)&Bt[(size_t)(n0 + r0) * K + sg * 8];
    rb1 = *(const uint4*)&Bt[(size_t)(n0 + r1) * K + sg * 8];

    for (int i = 0; i < nsteps; ++i) {
        __syncthreads();                       // LDS free to overwrite
        *(uint4*)&As[0][ldsOff]  = ra0;
        *(uint4*)&As[0][ldsOff2] = ra1;
        *(uint4*)&Bs[0][ldsOff]  = rb0;
        *(uint4*)&Bs[0][ldsOff2] = rb1;
        __syncthreads();                       // LDS filled

        if (i + 1 < nsteps) {
            int kk = (i + 1) << 5;
            ra0 = *(const uint4*)&A[(size_t)(m0 + r0) * K + kk + sg * 8];
            ra1 = *(const uint4*)&A[(size_t)(m0 + r1) * K + kk + sg * 8];
            rb0 = *(const uint4*)&Bt[(size_t)(n0 + r0) * K + kk + sg * 8];
            rb1 = *(const uint4*)&Bt[(size_t)(n0 + r1) * K + kk + sg * 8];
        }

        FragAB afr[2], bfr[4];
        #pragma unroll
        for (int im = 0; im < 2; ++im) {
            int row = waveM * 32 + im * 16 + l15;
            afr[im].q[0] = *(const uint4*)&As[0][row * LSTR + lhalf * 8];
            afr[im].q[1] = *(const uint4*)&As[0][row * LSTR + 16 + lhalf * 8];
        }
        #pragma unroll
        for (int in = 0; in < 4; ++in) {
            int col = waveN * 64 + in * 16 + l15;
            bfr[in].q[0] = *(const uint4*)&Bs[0][col * LSTR + lhalf * 8];
            bfr[in].q[1] = *(const uint4*)&Bs[0][col * LSTR + 16 + lhalf * 8];
        }
        #pragma unroll
        for (int im = 0; im < 2; ++im)
            #pragma unroll
            for (int in = 0; in < 4; ++in)
                acc[im][in] = __builtin_amdgcn_wmma_f32_16x16x32_bf16(
                    false, afr[im].v, false, bfr[in].v,
                    (short)0, acc[im][in], false, false);
    }
#endif

    // --- epilogue: C element r -> (row = 8*lhalf + r, col = l15) within a tile
    #pragma unroll
    for (int im = 0; im < 2; ++im) {
        #pragma unroll
        for (int in = 0; in < 4; ++in) {
            int gm = m0 + waveM * 32 + im * 16 + lhalf * 8;
            int gn = n0 + waveN * 64 + in * 16 + l15;
            float bb = addBias ? bias[gn] : 0.0f;
            #pragma unroll
            for (int r = 0; r < 8; ++r)
                C[(size_t)(gm + r) * N + gn] = acc[im][in][r] + bb;
        }
    }
}

// ---------------------------------------------------------------------------
// 4) Row-wise L2 normalize E (f32, 8192x512) -> En (bf16). One wave per row.
// ---------------------------------------------------------------------------
__global__ __launch_bounds__(256)
void normalize_kernel(const float* __restrict__ E, unsigned short* __restrict__ En) {
    int wid  = threadIdx.x >> 5;
    int lane = threadIdx.x & 31;
    int row  = blockIdx.x * 8 + wid;
    const float* e = E + (size_t)row * DDIM;

    float s = 0.0f;
    for (int j = lane; j < DDIM; j += 32) { float v = e[j]; s += v * v; }
    #pragma unroll
    for (int off = 16; off > 0; off >>= 1) s += __shfl_xor(s, off, 32);

    float scale = 1.0f / fmaxf(sqrtf(s), 1e-12f);
    unsigned short* o = En + (size_t)row * DDIM;
    for (int j = lane; j < DDIM; j += 32) o[j] = f2bf(e[j] * scale);
}

// ---------------------------------------------------------------------------
// 6) Per-row top-49 (diagonal excluded). One block per row; row cached in LDS.
// ---------------------------------------------------------------------------
__global__ __launch_bounds__(256)
void topk_kernel(const float* __restrict__ cosm,
                 float* __restrict__ vals, int* __restrict__ idxs) {
    __shared__ float srow[NROW];
    __shared__ float redV[256];
    __shared__ int   redI[256];

    const int tid = threadIdx.x;
    const int row = blockIdx.x;
    const float* src = cosm + (size_t)row * NROW;

    for (int j = tid; j < NROW; j += 256) srow[j] = src[j];
    __syncthreads();
    if (tid == 0) srow[row] = -INFINITY;      // exclude self
    __syncthreads();

    for (int t = 0; t < KTOP; ++t) {
        float best = -INFINITY; int bi = 0;
        for (int j = tid; j < NROW; j += 256) {
            float v = srow[j];
            if (v > best) { best = v; bi = j; }
        }
        redV[tid] = best; redI[tid] = bi;
        __syncthreads();
        #pragma unroll
        for (int s = 128; s > 0; s >>= 1) {
            if (tid < s && redV[tid + s] > redV[tid]) {
                redV[tid] = redV[tid + s]; redI[tid] = redI[tid + s];
            }
            __syncthreads();
        }
        if (tid == 0) {
            vals[(size_t)row * KTOP + t] = redV[0];
            idxs[(size_t)row * KTOP + t] = redI[0];
            srow[redI[0]] = -INFINITY;
        }
        __syncthreads();
    }
}

// ---------------------------------------------------------------------------
// 7) Zero output
// ---------------------------------------------------------------------------
__global__ __launch_bounds__(256)
void zero_kernel(float4* __restrict__ p, long n4) {
    long stride = (long)gridDim.x * 256;
    for (long i = (long)blockIdx.x * 256 + threadIdx.x; i < n4; i += stride)
        p[i] = (float4){0.f, 0.f, 0.f, 0.f};
}

// ---------------------------------------------------------------------------
// 8) Scatter top-k symmetrically + unit diagonal
// ---------------------------------------------------------------------------
__global__ __launch_bounds__(256)
void scatter_kernel(const float* __restrict__ vals, const int* __restrict__ idxs,
                    float* __restrict__ S) {
    int gid = blockIdx.x * 256 + threadIdx.x;
    if (gid < NROW * KTOP) {
        int r = gid / KTOP;
        int c = idxs[gid];
        float v = vals[gid];
        S[(size_t)r * NROW + c] = v;
        S[(size_t)c * NROW + r] = v;
    }
    if (gid < NROW) S[(size_t)gid * NROW + gid] = 1.0f;
}

// ---------------------------------------------------------------------------
extern "C" void kernel_launch(void* const* d_in, const int* in_sizes, int n_in,
                              void* d_out, int out_size, void* d_ws, size_t ws_size,
                              hipStream_t stream) {
    const float* X  = (const float*)d_in[0];   // 8192x512
    const float* sw = (const float*)d_in[1];   // 4
    const float* W0 = (const float*)d_in[2];
    const float* b0 = (const float*)d_in[3];
    const float* W1 = (const float*)d_in[4];
    const float* b1 = (const float*)d_in[5];
    const float* W2 = (const float*)d_in[6];
    const float* b2 = (const float*)d_in[7];
    const float* W3 = (const float*)d_in[8];
    const float* b3 = (const float*)d_in[9];
    float* S = (float*)d_out;

    // Workspace layout (all 16B-aligned offsets)
    char* w = (char*)d_ws;
    unsigned short* Wf  = (unsigned short*)(w);                              // 512KB
    float*          bf_ = (float*)         (w + 524288);                     // 2KB
    unsigned short* Xb  = (unsigned short*)(w + 526336);                     // 8MB
    float*          E   = (float*)         (w + 8914944);                    // 16MB
    unsigned short* En  = (unsigned short*)(w + 25692160);                   // 8MB
    float*          tv  = (float*)         (w + 34080768);                   // 1.6MB
    int*            ti  = (int*)           (w + 35686400);                   // 1.6MB

    // 1) blend weights
    fuse_weights_kernel<<<(DDIM * DDIM + 255) / 256, 256, 0, stream>>>(
        sw, W0, b0, W1, b1, W2, b2, W3, b3, Wf, bf_);

    // 2) X -> bf16
    f32_to_bf16_kernel<<<(NROW * DDIM) / 256, 256, 0, stream>>>(
        X, Xb, (long)NROW * DDIM);

    // 3) E = X * Wf^T + b     (M=8192, N=512, K=512)
    wmma_gemm_abt_kernel<<<dim3(DDIM / 128, NROW / 128), 256, 0, stream>>>(
        Xb, Wf, bf_, E, NROW, DDIM, DDIM, 1);

    // 4) row L2 normalize -> bf16
    normalize_kernel<<<NROW / 8, 256, 0, stream>>>(E, En);

    // 5) cos = En * En^T  -> d_out   (M=N=8192, K=512)
    wmma_gemm_abt_kernel<<<dim3(NROW / 128, NROW / 128), 256, 0, stream>>>(
        En, En, bf_, S, NROW, NROW, DDIM, 0);

    // 6) per-row top-49
    topk_kernel<<<NROW, 256, 0, stream>>>(S, tv, ti);

    // 7) zero S
    zero_kernel<<<16384, 256, 0, stream>>>((float4*)S, (long)NROW * NROW / 4);

    // 8) scatter symmetric values + diagonal
    scatter_kernel<<<(NROW * KTOP + 255) / 256, 256, 0, stream>>>(tv, ti, S);
}
